// DifBlock_63565515981221
// MI455X (gfx1250) — matrix operand order repaired and checked by reading, not compiled, not run on hardware
//
#include <hip/hip_runtime.h>

// ---------------------------------------------------------------------------
// DifBlock for MI455X (gfx1250): bf16 WMMA 16x16x32, f32 accumulate.
// B operands pre-packed into WMMA register order (32B contiguous per lane)
// so the GEMM inner loops issue only b128-class loads.
// Shapes: B=4, L=12, N=2048, D=64, K_T=3, FK=256, t0=10, 11 AR steps
// ---------------------------------------------------------------------------

typedef __attribute__((ext_vector_type(16))) __bf16 v16bf;
typedef __attribute__((ext_vector_type(8)))  float  v8f;

#define NN   2048
#define DD   64
#define BB_  4
#define LL   12
#define TT0  10
#define KFC  192
#define FK_  256

__device__ __forceinline__ v8f wmma_bf(v16bf a, v16bf b, v8f c) {
  return __builtin_amdgcn_wmma_f32_16x16x32_bf16(false, a, false, b, (short)0, c,
                                                 false, false);
}

// A-tile 16x32 (MxK) from row-major bf16; layout used here (matches packed B): lane<16 holds row M=lane,
// K in {0..15}; lane>=16 same row, K in {16..31}.
__device__ __forceinline__ v16bf load_a_bf(const __bf16* __restrict__ p, int ld) {
  int l = threadIdx.x & 31;
  const __bf16* rp = p + (l & 15) * ld + ((l >> 4) << 4);
  v16bf a;
#pragma unroll
  for (int e = 0; e < 16; ++e) a[e] = rp[e];
  return a;
}

__device__ __forceinline__ v16bf load_a_f32(const float* __restrict__ p, int ld) {
  int l = threadIdx.x & 31;
  const float* rp = p + (l & 15) * ld + ((l >> 4) << 4);
  v16bf a;
#pragma unroll
  for (int e = 0; e < 16; ++e) a[e] = (__bf16)rp[e];
  return a;
}

// B-tile from pre-packed panel: 512 bf16 per 32x16 tile, lane l owns 16
// contiguous bf16 at panel + l*16  -> one 32-byte vector load.
__device__ __forceinline__ v16bf load_b_packed(const __bf16* __restrict__ panel) {
  int l = threadIdx.x & 31;
  return *(const v16bf*)(panel + (l << 4));
}

// ---------------------------------------------------------------------------
__global__ void k_cvt_bf16(const float* __restrict__ s, __bf16* __restrict__ d, int n) {
  int i = blockIdx.x * blockDim.x + threadIdx.x;
  if (i < n) d[i] = (__bf16)s[i];
}

// Pack row-major f32 (K x N, leading dim ld) into WMMA-B panels.
// Panel p = kt*ntiles + nt holds the 32x16 tile; element (k,n) of the tile goes
// to lane = (n&15) + 16*((k>>4)&1), slot e = k&15. One thread per (panel,lane).
// zero_diag: zero elements with global k == global n (for g1 = G .* (1-I)).
__global__ void k_pack_b(const float* __restrict__ src, long src_sstride, int ld,
                         int ktiles, int ntiles, int zero_diag,
                         __bf16* __restrict__ dst, long dst_sstride) {
  int idx = blockIdx.x * blockDim.x + threadIdx.x;
  if (idx >= ktiles * ntiles * 32) return;
  src += (long)blockIdx.y * src_sstride;
  dst += (long)blockIdx.y * dst_sstride;
  int lane = idx & 31;
  int p = idx >> 5;
  int nt = p % ntiles, kt = p / ntiles;
  int col  = nt * 16 + (lane & 15);
  int krow = kt * 32 + ((lane >> 4) << 4);
  const float* s = src + (long)krow * ld + col;
  v16bf tmp;
#pragma unroll
  for (int e = 0; e < 16; ++e) {
    float v = s[(long)e * ld];
    if (zero_diag && (krow + e) == col) v = 0.f;
    tmp[e] = (__bf16)v;
  }
  *(v16bf*)(dst + (long)p * 512 + (lane << 4)) = tmp;
}

// Gsum = mask .* (G @ g1) + g1   (stored row-major bf16, used as A later).
// Wave: 16 rows x 64 cols, K=2048. B from packed g1 panels (128 ntiles).
__global__ void k_g2(const __bf16* __restrict__ Gb, const __bf16* __restrict__ g1p,
                     const float* __restrict__ G, __bf16* __restrict__ Gsum) {
  int n0 = blockIdx.y * 16;
  int c0 = blockIdx.x * 64;
  v8f acc[4] = {};
  for (int k0 = 0; k0 < NN; k0 += 32) {
    v16bf a = load_a_bf(Gb + n0 * NN + k0, NN);
    const __bf16* pp = g1p + ((long)((k0 >> 5) * 128 + (c0 >> 4)) << 9);
#pragma unroll
    for (int j = 0; j < 4; ++j)
      acc[j] = wmma_bf(a, load_b_packed(pp + ((long)j << 9)), acc[j]);
  }
  int l = threadIdx.x & 31;
  int col = l & 15, rb = (l >> 4) << 3;
#pragma unroll
  for (int j = 0; j < 4; ++j)
#pragma unroll
    for (int r = 0; r < 8; ++r) {
      int rr = n0 + rb + r, cc = c0 + j * 16 + col;
      float g = G[rr * NN + cc];
      float v = acc[j][r] + g;
      Gsum[rr * NN + cc] = (__bf16)((rr == cc) ? 0.f : v);
    }
}

// Xs = sum_{s<3} relu(Xw @ W_fc)[:, 64s:64s+64]   (X0 = Xs/3, fused later)
// One wave: 16 n-rows, all 12 N-tiles (192 cols), K=192 in 6 k-tiles.
__global__ void k_fc(const float* __restrict__ Xin, int Lsrc, int tt0, int t,
                     const __bf16* __restrict__ Wfcp, float* __restrict__ Xs) {
  int bt = blockIdx.x;                 // b*t slab
  int bb = bt / t, tt = bt % t;
  int n0 = blockIdx.y * 16;
  v8f acc[12] = {};
#pragma unroll
  for (int kt = 0; kt < 6; ++kt) {
    int slice = kt >> 1, db = (kt & 1) * 32;
    const float* ap = Xin + (((bb * Lsrc + tt0 + tt + slice) * NN) + n0) * DD + db;
    v16bf a = load_a_f32(ap, DD);
    const __bf16* pp = Wfcp + ((long)(kt * 12) << 9);
#pragma unroll
    for (int j = 0; j < 12; ++j)
      acc[j] = wmma_bf(a, load_b_packed(pp + ((long)j << 9)), acc[j]);
  }
  float* out = Xs + (bt * NN + n0) * DD;
  int l = threadIdx.x & 31;
  int col = l & 15, rb = (l >> 4) << 3;
#pragma unroll
  for (int j = 0; j < 4; ++j)
#pragma unroll
    for (int r = 0; r < 8; ++r) {
      float v = fmaxf(acc[j][r], 0.f) + fmaxf(acc[j + 4][r], 0.f) +
                fmaxf(acc[j + 8][r], 0.f);
      out[(rb + r) * DD + j * 16 + col] = v;
    }
}

// Y = Xs/3 + Gsum @ Xs   per (b,t) slab. Wave: 16 rows x 64 cols, K=2048.
// B from per-slab packed Xs panels (64 ktiles x 4 ntiles).
__global__ void k_mix(const __bf16* __restrict__ Gsum, const float* __restrict__ Xs,
                      const __bf16* __restrict__ Xsp, float* __restrict__ Y) {
  int slab = blockIdx.x;
  int n0 = blockIdx.y * 16;
  const __bf16* xpp = Xsp + (long)slab * (64 * 4 * 512);
  v8f acc[4] = {};
  for (int k0 = 0; k0 < NN; k0 += 32) {
    v16bf a = load_a_bf(Gsum + n0 * NN + k0, NN);
    const __bf16* pp = xpp + ((long)((k0 >> 5) * 4) << 9);
#pragma unroll
    for (int j = 0; j < 4; ++j)
      acc[j] = wmma_bf(a, load_b_packed(pp + ((long)j << 9)), acc[j]);
  }
  float* out = Y + (long)slab * NN * DD + n0 * DD;
  const float* xrow = Xs + (long)slab * NN * DD + n0 * DD;
  int l = threadIdx.x & 31;
  int col = l & 15, rb = (l >> 4) << 3;
#pragma unroll
  for (int j = 0; j < 4; ++j)
#pragma unroll
    for (int r = 0; r < 8; ++r) {
      int rr = rb + r, cc = j * 16 + col;
      out[rr * DD + cc] = acc[j][r] + xrow[rr * DD + cc] * (1.f / 3.f);
    }
}

// dst = A @ W(64x64, packed panels 2x4) + bias ; A slabs contiguous (NN x 64 f32)
__global__ void k_gemm64(const float* __restrict__ A, const __bf16* __restrict__ Wp,
                         const float* __restrict__ bias, float* __restrict__ dst,
                         long dst_slab_stride) {
  int slab = blockIdx.x;
  int n0 = blockIdx.y * 16;
  const float* ap = A + (long)slab * NN * DD + n0 * DD;
  v8f acc[4] = {};
#pragma unroll
  for (int kt = 0; kt < 2; ++kt) {
    v16bf a = load_a_f32(ap + kt * 32, DD);
    const __bf16* pp = Wp + ((long)(kt * 4) << 9);
#pragma unroll
    for (int j = 0; j < 4; ++j)
      acc[j] = wmma_bf(a, load_b_packed(pp + ((long)j << 9)), acc[j]);
  }
  float* out = dst + slab * dst_slab_stride + n0 * DD;
  int l = threadIdx.x & 31;
  int col = l & 15, rb = (l >> 4) << 3;
#pragma unroll
  for (int j = 0; j < 4; ++j) {
    float bv = bias[j * 16 + col];
#pragma unroll
    for (int r = 0; r < 8; ++r)
      out[(rb + r) * DD + j * 16 + col] = acc[j][r] + bv;
  }
}

// E[:,0:2] = X_spa[:,10:12];  E[:,2] = Z[:,9]
__global__ void k_build_e(const float* __restrict__ Xspa, const float* __restrict__ Z,
                          float* __restrict__ E) {
  int i = blockIdx.x * blockDim.x + threadIdx.x;
  if (i >= BB_ * 3 * NN * DD) return;
  int d = i & (DD - 1);
  int n = (i >> 6) & (NN - 1);
  int slot = (i / (NN * DD)) % 3;
  int bb = i / (3 * NN * DD);
  float v = (slot < 2) ? Xspa[(((bb * LL) + 10 + slot) * NN + n) * DD + d]
                       : Z[(((bb * TT0) + 9) * NN + n) * DD + d];
  E[(((bb * 14) + slot) * NN + n) * DD + d] = v;
}

// forecast = P @ W_fore(64x256, packed 2x16 panels) + b_fore ; P = E[:,2:14].
// 4 waves/block, each owning a 64-col group.
__global__ void k_fore(const float* __restrict__ E, const __bf16* __restrict__ Wfp,
                       const float* __restrict__ bf_, float* __restrict__ out) {
  int s = blockIdx.x;                 // 0..47 = (b,tt)
  int bb = s / 12, tt = s % 12;
  int n0 = blockIdx.y * 16;
  int w = threadIdx.x >> 5;           // col group of 64
  const float* ap = E + (((bb * 14) + 2 + tt) * NN + n0) * DD;
  v8f acc[4] = {};
#pragma unroll
  for (int kt = 0; kt < 2; ++kt) {
    v16bf a = load_a_f32(ap + kt * 32, DD);
    const __bf16* pp = Wfp + ((long)(kt * 16 + w * 4) << 9);
#pragma unroll
    for (int j = 0; j < 4; ++j)
      acc[j] = wmma_bf(a, load_b_packed(pp + ((long)j << 9)), acc[j]);
  }
  float* op = out + (long)(s * NN + n0) * FK_ + w * 64;
  int l = threadIdx.x & 31;
  int col = l & 15, rb = (l >> 4) << 3;
#pragma unroll
  for (int j = 0; j < 4; ++j) {
    float bv = bf_[w * 64 + j * 16 + col];
#pragma unroll
    for (int r = 0; r < 8; ++r)
      op[(rb + r) * FK_ + j * 16 + col] = acc[j][r] + bv;
  }
}

// u = X[:,2:12] - relu(back); LayerNorm over D=64. 16 lanes per row.
__global__ void k_ln(const float* __restrict__ X, const float* __restrict__ back,
                     const float* __restrict__ gamma, const float* __restrict__ beta,
                     float* __restrict__ out) {
  int row = blockIdx.x * (blockDim.x >> 4) + (threadIdx.x >> 4);
  int sub = threadIdx.x & 15;
  if (row >= BB_ * TT0 * NN) return;
  int n = row & (NN - 1);
  int tt = (row >> 11) % TT0;
  int bb = row / (TT0 * NN);
  const float4 xv = *(const float4*)(X + (((bb * LL) + 2 + tt) * NN + n) * DD + sub * 4);
  const float4 bv = *(const float4*)(back + (long)row * DD + sub * 4);
  float u0 = xv.x - fmaxf(bv.x, 0.f);
  float u1 = xv.y - fmaxf(bv.y, 0.f);
  float u2 = xv.z - fmaxf(bv.z, 0.f);
  float u3 = xv.w - fmaxf(bv.w, 0.f);
  float s  = u0 + u1 + u2 + u3;
  float sq = u0 * u0 + u1 * u1 + u2 * u2 + u3 * u3;
#pragma unroll
  for (int m = 1; m < 16; m <<= 1) {
    s  += __shfl_xor(s,  m, 16);
    sq += __shfl_xor(sq, m, 16);
  }
  float mu  = s * (1.f / 64.f);
  float var = sq * (1.f / 64.f) - mu * mu;
  float inv = rsqrtf(var + 1e-5f);
  float* op = out + (long)row * DD + sub * 4;
  const float* gp = gamma + sub * 4;
  const float* bp = beta + sub * 4;
  op[0] = (u0 - mu) * inv * gp[0] + bp[0];
  op[1] = (u1 - mu) * inv * gp[1] + bp[1];
  op[2] = (u2 - mu) * inv * gp[2] + bp[2];
  op[3] = (u3 - mu) * inv * gp[3] + bp[3];
}

// ---------------------------------------------------------------------------
extern "C" void kernel_launch(void* const* d_in, const int* in_sizes, int n_in,
                              void* d_out, int out_size, void* d_ws, size_t ws_size,
                              hipStream_t stream) {
  (void)in_sizes; (void)n_in; (void)out_size; (void)ws_size;
  const float* X     = (const float*)d_in[0];
  const float* Xspa  = (const float*)d_in[1];
  const float* G     = (const float*)d_in[2];
  const float* Wfc   = (const float*)d_in[3];
  const float* Wgcn  = (const float*)d_in[4];
  const float* bgcn  = (const float*)d_in[5];
  const float* Wfore = (const float*)d_in[6];
  const float* bfore = (const float*)d_in[7];
  const float* Wback = (const float*)d_in[8];
  const float* bback = (const float*)d_in[9];
  const float* gam   = (const float*)d_in[10];
  const float* bet   = (const float*)d_in[11];

  char* w = (char*)d_ws;
  auto alloc = [&](size_t bytes) -> char* {
    char* p = w; w += (bytes + 255) & ~(size_t)255; return p;
  };
  __bf16* Gb      = (__bf16*)alloc((size_t)NN * NN * 2);           // G as A, bf16
  __bf16* g1p     = (__bf16*)alloc((size_t)NN * NN * 2);           // g1 B-panels
  __bf16* Gsb     = (__bf16*)alloc((size_t)NN * NN * 2);           // Gsum (A)
  __bf16* Wfc_p   = (__bf16*)alloc((size_t)KFC * KFC * 2);         // 6x12 panels
  __bf16* Wgcn_p  = (__bf16*)alloc((size_t)DD * DD * 2);           // 2x4 panels
  __bf16* Wfore_p = (__bf16*)alloc((size_t)DD * FK_ * 2);          // 2x16 panels
  __bf16* Wback_p = (__bf16*)alloc((size_t)DD * DD * 2);           // 2x4 panels
  float*  Xs      = (float*)alloc((size_t)BB_ * TT0 * NN * DD * 4);  // + back_buf
  __bf16* Xs_p    = (__bf16*)alloc((size_t)BB_ * TT0 * NN * DD * 2); // B-panels
  float*  Y       = (float*)alloc((size_t)BB_ * TT0 * NN * DD * 4);
  float*  Z       = (float*)alloc((size_t)BB_ * TT0 * NN * DD * 4);
  float*  E       = (float*)alloc((size_t)BB_ * 14 * NN * DD * 4);
  float*  backb   = Xs;  // reuse: Xs dead before backcast GEMM

  const long SLAB = (long)NN * DD;   // elements per (b,t) slab

  // graph + weight prep
  k_cvt_bf16<<<(NN * NN) / 256, 256, 0, stream>>>(G, Gb, NN * NN);
  k_pack_b<<<dim3((64 * 128 * 32) / 256, 1), 256, 0, stream>>>(
      G, 0, NN, 64, 128, 1, g1p, 0);
  k_pack_b<<<dim3((6 * 12 * 32 + 255) / 256, 1), 256, 0, stream>>>(
      Wfc, 0, KFC, 6, 12, 0, Wfc_p, 0);
  k_pack_b<<<dim3(1, 1), 256, 0, stream>>>(Wgcn, 0, DD, 2, 4, 0, Wgcn_p, 0);
  k_pack_b<<<dim3((2 * 16 * 32 + 255) / 256, 1), 256, 0, stream>>>(
      Wfore, 0, FK_, 2, 16, 0, Wfore_p, 0);
  k_pack_b<<<dim3(1, 1), 256, 0, stream>>>(Wback, 0, DD, 2, 4, 0, Wback_p, 0);

  // Gsum = g1 + mask.*(G @ g1)
  k_g2<<<dim3(NN / 64, NN / 16), 32, 0, stream>>>(Gb, g1p, G, Gsb);

  // Z = st_conv(X_spa)  (t = 10)
  k_fc<<<dim3(BB_ * TT0, NN / 16), 32, 0, stream>>>(Xspa, LL, 0, TT0, Wfc_p, Xs);
  k_pack_b<<<dim3((64 * 4 * 32) / 256, BB_ * TT0), 256, 0, stream>>>(
      Xs, SLAB, DD, 64, 4, 0, Xs_p, SLAB);
  k_mix<<<dim3(BB_ * TT0, NN / 16), 32, 0, stream>>>(Gsb, Xs, Xs_p, Y);
  k_gemm64<<<dim3(BB_ * TT0, NN / 16), 32, 0, stream>>>(Y, Wgcn_p, bgcn, Z, SLAB);

  // E = [X_spa[:,10:12], Z[:,9], P1..P11]
  k_build_e<<<(BB_ * 3 * NN * DD + 255) / 256, 256, 0, stream>>>(Xspa, Z, E);

  // 11 sequential autoregressive steps: window E[:, s-1:s+2] -> E[:, 2+s]
  for (int s = 1; s <= 11; ++s) {
    k_fc<<<dim3(BB_, NN / 16), 32, 0, stream>>>(E, 14, s - 1, 1, Wfc_p, Xs);
    k_pack_b<<<dim3((64 * 4 * 32) / 256, BB_), 256, 0, stream>>>(
        Xs, SLAB, DD, 64, 4, 0, Xs_p, SLAB);
    k_mix<<<dim3(BB_, NN / 16), 32, 0, stream>>>(Gsb, Xs, Xs_p, Y);
    k_gemm64<<<dim3(BB_, NN / 16), 32, 0, stream>>>(
        Y, Wgcn_p, bgcn, E + (size_t)(2 + s) * SLAB, (long)14 * SLAB);
  }

  float* out_back = (float*)d_out;
  float* out_fore = out_back + (size_t)BB_ * TT0 * NN * DD;

  // forecast = P @ W_fore + b_fore
  k_fore<<<dim3(BB_ * 12, NN / 16), 128, 0, stream>>>(E, Wfore_p, bfore, out_fore);

  // backcast: back = Z @ W_back + b_back, then u = Xr - relu(back), LayerNorm
  k_gemm64<<<dim3(BB_ * TT0, NN / 16), 32, 0, stream>>>(Z, Wback_p, bback, backb, SLAB);
  k_ln<<<(BB_ * TT0 * NN) / 16, 256, 0, stream>>>(X, backb, gam, bet, out_back);
}
